// MambaLayer_23905787969605
// MI455X (gfx1250) — matrix operand order repaired and checked by reading, not compiled, not run on hardware
//
#include <hip/hip_runtime.h>

typedef unsigned short u16;
typedef __bf16 bf16_t;
typedef bf16_t v16bf __attribute__((ext_vector_type(16)));
typedef float  v8f   __attribute__((ext_vector_type(8)));

#define D_MODEL   1024
#define D_STATE   128
#define D_CONV    4
#define D_INNER   2048
#define NHEADS    32
#define HEADDIM   64
#define CONV_DIM  2304
#define D_IN_PROJ 6176
#define CHUNK     64
#define BATCH     4
#define SEQ       2048
#define ROWS      (BATCH*SEQ)      // 8192
#define NCHUNK    (SEQ/CHUNK)      // 32
#define EPSF      1e-5f

// ---------- bf16 helpers ----------
__device__ __forceinline__ u16 f2bf(float f) {
  union { float f; unsigned u; } a; a.f = f;
  unsigned r = a.u + 0x7FFFu + ((a.u >> 16) & 1u);
  return (u16)(r >> 16);
}
__device__ __forceinline__ float bf2f(u16 h) {
  union { unsigned u; float f; } a; a.u = ((unsigned)h) << 16;
  return a.f;
}

union Frag16 {
  v16bf bf;
  u16 u[16];
  unsigned d[8];
};

__device__ __forceinline__ v8f wmma_bf16(v16bf a, v16bf b, v8f c) {
  // D = A(16x32) * B(32x16) + C  -> v_wmma_f32_16x16x32_bf16
  return __builtin_amdgcn_wmma_f32_16x16x32_bf16(
      false, a, false, b, (short)0, c, false, false);
}

// A fragment, row-major global tile (element (m,k) at tile[m*ld+k]); 16B vector loads.
__device__ __forceinline__ v16bf gfrag_a(const u16* tile, int ld) {
  int lane = threadIdx.x & 31;
  int m = lane & 15, kh = lane >> 4;
  const unsigned* p0 = (const unsigned*)(tile + (size_t)m * ld + kh * 16);
  const unsigned* p1 = (const unsigned*)(tile + (size_t)m * ld + kh * 16 + 8);
  Frag16 f;
#pragma unroll
  for (int i = 0; i < 4; ++i) { f.d[i] = p0[i]; f.d[4 + i] = p1[i]; }
  return f.bf;
}
// B fragment where memory holds B^T row-major (i.e. W[n][k]); column n contiguous in k.
__device__ __forceinline__ v16bf gfrag_b(const u16* tile, int ld) {
  int lane = threadIdx.x & 31;
  int n = lane & 15, kb = (lane >> 4) * 16;
  const unsigned* p = (const unsigned*)(tile + (size_t)n * ld + kb);
  Frag16 f;
#pragma unroll
  for (int i = 0; i < 8; ++i) f.d[i] = p[i];
  return f.bf;
}
// Generic strided fragment loaders (LDS tiles)
__device__ __forceinline__ v16bf frag_a_gen(const u16* base, int ms, int ks) {
  int lane = threadIdx.x & 31;
  int m = lane & 15, kh = lane >> 4;
  Frag16 f;
#pragma unroll
  for (int e = 0; e < 8; ++e) {
    f.u[e]     = base[m * ms + (kh * 16 + e) * ks];
    f.u[8 + e] = base[m * ms + (kh * 16 + 8 + e) * ks];
  }
  return f.bf;
}
__device__ __forceinline__ v16bf frag_b_gen(const u16* base, int ks, int ns) {
  int lane = threadIdx.x & 31;
  int n = lane & 15, kb = (lane >> 4) * 16;
  Frag16 f;
#pragma unroll
  for (int e = 0; e < 16; ++e) f.u[e] = base[(kb + e) * ks + n * ns];
  return f.bf;
}

// ---------- workspace layout (bytes) ----------
static const size_t OFF_XN = 0;                                   // 8192x1024 bf16
static const size_t OFF_WI = OFF_XN + (size_t)ROWS * D_MODEL * 2; // 6176x1024 bf16
static const size_t OFF_WO = OFF_WI + (size_t)D_IN_PROJ * D_MODEL * 2;
static const size_t OFF_ZX = OFF_WO + (size_t)D_MODEL * D_INNER * 2;     // f32 8192x6176
static const size_t OFF_DT = OFF_ZX + (size_t)ROWS * D_IN_PROJ * 4;      // f32 8192x32
static const size_t OFF_AC = OFF_DT + (size_t)ROWS * NHEADS * 4;         // f32 4*32*32*64
static const size_t OFF_AL = OFF_AC + (size_t)BATCH * NCHUNK * NHEADS * CHUNK * 4;
static const size_t OFF_XS = OFF_AL + (size_t)BATCH * NCHUNK * NHEADS * 4;
static const size_t OFF_XD = OFF_XS + (size_t)ROWS * D_INNER * 2;
static const size_t OFF_BM = OFF_XD + (size_t)ROWS * D_INNER * 2;
static const size_t OFF_CM = OFF_BM + (size_t)ROWS * D_STATE * 2;
static const size_t OFF_SL = OFF_CM + (size_t)ROWS * D_STATE * 2;        // bf16 states
static const size_t OFF_SP = OFF_SL + (size_t)BATCH * NCHUNK * NHEADS * HEADDIM * D_STATE * 2;
static const size_t OFF_Y  = OFF_SP + (size_t)BATCH * NCHUNK * NHEADS * HEADDIM * D_STATE * 2;
static const size_t OFF_YG = OFF_Y  + (size_t)ROWS * D_INNER * 4;

// ---------- kernels ----------
__global__ __launch_bounds__(256) void k_rmsnorm_cast(const float* __restrict__ x,
                                                      const float* __restrict__ w,
                                                      u16* __restrict__ xn) {
  int row = blockIdx.x;
  const float* xr = x + (size_t)row * D_MODEL;
  __shared__ float red[256];
  float ss = 0.f;
  for (int i = threadIdx.x; i < D_MODEL; i += 256) { float v = xr[i]; ss += v * v; }
  red[threadIdx.x] = ss; __syncthreads();
  for (int s = 128; s > 0; s >>= 1) {
    if (threadIdx.x < s) red[threadIdx.x] += red[threadIdx.x + s];
    __syncthreads();
  }
  float sc = rsqrtf(red[0] / (float)D_MODEL + EPSF);
  for (int i = threadIdx.x; i < D_MODEL; i += 256)
    xn[(size_t)row * D_MODEL + i] = f2bf(xr[i] * sc * w[i]);
}

__global__ __launch_bounds__(256) void k_cast(const float* __restrict__ s,
                                              u16* __restrict__ d, int n) {
  int i = blockIdx.x * 256 + threadIdx.x;
  if (i < n) d[i] = f2bf(s[i]);
}

// C[M,N] = A[M,K](bf16) * W[N,K]^T(bf16) (+resid).
// Block: 8 waves stacked on M, block tile 256(M) x 32(N).
// Each wave: 2x2 register tile (32x32) -> 4 WMMA per K-step, frags reused.
__global__ __launch_bounds__(256) void k_gemm(const u16* __restrict__ A,
                                              const u16* __restrict__ W,
                                              float* __restrict__ C,
                                              const float* __restrict__ resid,
                                              int M, int N, int K) {
  int lane = threadIdx.x & 31;
  int w = threadIdx.x >> 5;
  int m0 = blockIdx.y * 256 + w * 32;
  int n0 = blockIdx.x * 32;
  const u16* Ab0 = A + (size_t)m0 * K;
  const u16* Ab1 = A + (size_t)(m0 + 16) * K;
  const u16* Wb0 = W + (size_t)n0 * K;
  const u16* Wb1 = W + (size_t)(n0 + 16) * K;
  v8f acc00 = {}, acc01 = {}, acc10 = {}, acc11 = {};
  for (int k0 = 0; k0 < K; k0 += 32) {
    __builtin_prefetch(Ab0 + k0 + 256, 0, 3);   // global_prefetch_b8
    __builtin_prefetch(Ab1 + k0 + 256, 0, 3);
    v16bf a0 = gfrag_a(Ab0 + k0, K);
    v16bf a1 = gfrag_a(Ab1 + k0, K);
    v16bf b0 = gfrag_b(Wb0 + k0, K);
    v16bf b1 = gfrag_b(Wb1 + k0, K);
    acc00 = wmma_bf16(a0, b0, acc00);
    acc01 = wmma_bf16(a0, b1, acc01);
    acc10 = wmma_bf16(a1, b0, acc10);
    acc11 = wmma_bf16(a1, b1, acc11);
  }
  int nn = lane & 15;
  int mh = (lane >> 4) * 8;
#pragma unroll
  for (int mi = 0; mi < 2; ++mi) {
#pragma unroll
    for (int ni = 0; ni < 2; ++ni) {
      v8f acc = (mi == 0) ? (ni == 0 ? acc00 : acc01)
                          : (ni == 0 ? acc10 : acc11);
      int n = n0 + ni * 16 + nn;
      int mb = m0 + mi * 16 + mh;
#pragma unroll
      for (int r = 0; r < 8; ++r) {
        size_t idx = (size_t)(mb + r) * N + n;
        float v = acc[r];
        if (resid) v += resid[idx];
        C[idx] = v;
      }
    }
  }
}

__global__ __launch_bounds__(256) void k_dt(const float* __restrict__ zx,
                                            const float* __restrict__ dt_bias,
                                            float* __restrict__ dt) {
  int i = blockIdx.x * 256 + threadIdx.x;
  if (i < ROWS * NHEADS) {
    int h = i & 31, row = i >> 5;
    float v = zx[(size_t)row * D_IN_PROJ + (D_INNER + CONV_DIM) + h] + dt_bias[h];
    dt[i] = (v > 20.f) ? v : log1pf(expf(v));
  }
}

__global__ __launch_bounds__(64) void k_cumsum(const float* __restrict__ dt,
                                               const float* __restrict__ A_log,
                                               float* __restrict__ Ac,
                                               float* __restrict__ Alast) {
  int h = blockIdx.x & 31;
  int bc = blockIdx.x >> 5;           // b*NCHUNK + c
  int l = threadIdx.x;
  float Ah = -expf(A_log[h]);
  float a = dt[(size_t)(bc * CHUNK + l) * NHEADS + h] * Ah;
  __shared__ float s[64];
  s[l] = a; __syncthreads();
  for (int off = 1; off < 64; off <<= 1) {
    float t = (l >= off) ? s[l - off] : 0.f;
    __syncthreads();
    s[l] += t;
    __syncthreads();
  }
  Ac[(size_t)blockIdx.x * CHUNK + l] = s[l];
  if (l == 63) Alast[blockIdx.x] = s[63];
}

__global__ __launch_bounds__(256) void k_conv(const float* __restrict__ zx,
                                              const float* __restrict__ conv_w,
                                              const float* __restrict__ conv_b,
                                              const float* __restrict__ dt,
                                              u16* __restrict__ xs, u16* __restrict__ xd,
                                              u16* __restrict__ Bm, u16* __restrict__ Cm) {
  int row = blockIdx.x;
  int b = row >> 11, l = row & 2047;
  for (int c = threadIdx.x; c < CONV_DIM; c += 256) {
    float acc = conv_b[c];
#pragma unroll
    for (int k = 0; k < D_CONV; ++k) {
      int ls = l - (D_CONV - 1) + k;
      if (ls >= 0)
        acc += zx[(size_t)(b * SEQ + ls) * D_IN_PROJ + D_INNER + c] * conv_w[k * CONV_DIM + c];
    }
    float v = acc / (1.f + expf(-acc));     // silu
    if (c < D_INNER) {
      xs[(size_t)row * D_INNER + c] = f2bf(v);
      int h = c >> 6;
      xd[(size_t)row * D_INNER + c] = f2bf(v * dt[(size_t)row * NHEADS + h]);
    } else if (c < D_INNER + D_STATE) {
      Bm[(size_t)row * D_STATE + (c - D_INNER)] = f2bf(v);
    } else {
      Cm[(size_t)row * D_STATE + (c - D_INNER - D_STATE)] = f2bf(v);
    }
  }
}

// One block per (b, chunk): G = C*B^T once, then per-head diag term + chunk states.
__global__ __launch_bounds__(256) void k_ssd_intra(const u16* __restrict__ Bm,
                                                   const u16* __restrict__ Cm,
                                                   const u16* __restrict__ Xd,
                                                   const u16* __restrict__ xs,
                                                   const float* __restrict__ Ac,
                                                   const float* __restrict__ Alast,
                                                   const float* __restrict__ Dp,
                                                   float* __restrict__ Y,
                                                   u16* __restrict__ Slocal) {
  __shared__ __align__(16) char smem[57600];
  u16*  sB  = (u16*)smem;              // [64][128]
  float* sG = (float*)(smem + 16384);  // [64][64]
  u16*  sC  = (u16*)(smem + 32768);    // [64][128] phase 1
  u16*  sM  = (u16*)(smem + 32768);    // [64][64]  phase 2
  u16*  sX  = (u16*)(smem + 40960);    // [64][64]
  u16*  sXdT= (u16*)(smem + 49152);    // [64][64]  (transposed, decayed)
  float* sAc= (float*)(smem + 57344);  // [64]
  int bc = blockIdx.x;
  int tid = threadIdx.x, lane = tid & 31, w = tid >> 5;
  size_t rowbase = (size_t)bc * CHUNK;

  for (int i = tid; i < CHUNK * D_STATE; i += 256) {
    sB[i] = Bm[rowbase * D_STATE + i];
    sC[i] = Cm[rowbase * D_STATE + i];
  }
  __syncthreads();

  // G[l][s] = sum_n C[l][n] B[s][n]
#pragma unroll
  for (int s2 = 0; s2 < 2; ++s2) {
    int st = w + s2 * 8, mt = st & 3, nt = st >> 2;
    v8f acc = {};
    for (int k0 = 0; k0 < D_STATE; k0 += 32) {
      v16bf af = frag_a_gen(sC + mt * 16 * D_STATE + k0, D_STATE, 1);
      v16bf bf = frag_b_gen(sB + nt * 16 * D_STATE + k0, 1, D_STATE);
      acc = wmma_bf16(af, bf, acc);
    }
    int n = nt * 16 + (lane & 15), mh = mt * 16 + (lane >> 4) * 8;
#pragma unroll
    for (int r = 0; r < 8; ++r) sG[(mh + r) * 64 + n] = acc[r];
  }
  __syncthreads();

  for (int h = 0; h < NHEADS; ++h) {
    if (tid < 64) sAc[tid] = Ac[((size_t)bc * NHEADS + h) * CHUNK + tid];
    __syncthreads();
    float alast = sAc[63];
    for (int i = tid; i < 64 * 64; i += 256) {
      int l = i >> 6, p = i & 63;
      u16 xv = Xd[(rowbase + l) * D_INNER + h * HEADDIM + p];
      sX[i] = xv;
      sXdT[p * 64 + l] = f2bf(expf(alast - sAc[l]) * bf2f(xv));
    }
    for (int i = tid; i < 64 * 64; i += 256) {
      int l = i >> 6, s = i & 63;
      float m = (l >= s) ? expf(sAc[l] - sAc[s]) * sG[i] : 0.f;
      sM[i] = f2bf(m);
    }
    __syncthreads();

    float Dh = Dp[h];
    // Yd = M * X  (+ D*xs)
#pragma unroll
    for (int s2 = 0; s2 < 2; ++s2) {
      int st = w + s2 * 8, mt = st & 3, nt = st >> 2;
      v8f acc = {};
      for (int k0 = 0; k0 < 64; k0 += 32) {
        v16bf af = frag_a_gen(sM + mt * 16 * 64 + k0, 64, 1);
        v16bf bf = frag_b_gen(sX + k0 * 64 + nt * 16, 64, 1);
        acc = wmma_bf16(af, bf, acc);
      }
      int n = nt * 16 + (lane & 15), mh = mt * 16 + (lane >> 4) * 8;
#pragma unroll
      for (int r = 0; r < 8; ++r) {
        int l = mh + r;
        size_t yi = ((rowbase + l) * NHEADS + h) * HEADDIM + n;
        float xv = bf2f(xs[(rowbase + l) * D_INNER + h * HEADDIM + n]);
        Y[yi] = acc[r] + Dh * xv;
      }
    }
    // S[p][n] = sum_l XdT[p][l] * B[l][n]
#pragma unroll
    for (int s2 = 0; s2 < 4; ++s2) {
      int st = w * 4 + s2, mt = st >> 3, nt = st & 7;
      v8f acc = {};
      for (int k0 = 0; k0 < 64; k0 += 32) {
        v16bf af = frag_a_gen(sXdT + mt * 16 * 64 + k0, 64, 1);
        v16bf bf = frag_b_gen(sB + k0 * D_STATE + nt * 16, D_STATE, 1);
        acc = wmma_bf16(af, bf, acc);
      }
      int n = nt * 16 + (lane & 15), mh = mt * 16 + (lane >> 4) * 8;
      size_t sb = ((size_t)bc * NHEADS + h) * (HEADDIM * D_STATE);
#pragma unroll
      for (int r = 0; r < 8; ++r)
        Slocal[sb + (size_t)(mh + r) * D_STATE + n] = f2bf(acc[r]);
    }
    __syncthreads();
  }
}

// Inter-chunk sequential scan, one block per (b,h)
__global__ __launch_bounds__(256) void k_ssd_scan(const u16* __restrict__ Slocal,
                                                  const float* __restrict__ Alast,
                                                  u16* __restrict__ Sprev) {
  int b = blockIdx.x >> 5, h = blockIdx.x & 31;
  int tid = threadIdx.x;
  float run[32];
#pragma unroll
  for (int i = 0; i < 32; ++i) run[i] = 0.f;
  for (int c = 0; c < NCHUNK; ++c) {
    size_t bc = (size_t)(b * NCHUNK + c);
    size_t base = (bc * NHEADS + h) * (HEADDIM * D_STATE);
    float a = expf(Alast[bc * NHEADS + h]);
#pragma unroll
    for (int i = 0; i < 32; ++i) {
      size_t idx = base + (size_t)i * 256 + tid;
      Sprev[idx] = f2bf(run[i]);
      run[i] = a * run[i] + bf2f(Slocal[idx]);
    }
  }
}

// Yo = exp(Ac[l]) * C * Sprev^T, accumulated into Y
__global__ __launch_bounds__(256) void k_ssd_inter(const u16* __restrict__ Cm,
                                                   const u16* __restrict__ Sprev,
                                                   const float* __restrict__ Ac,
                                                   float* __restrict__ Y) {
  __shared__ __align__(16) char smem[33024];
  u16* sC = (u16*)smem;               // [64][128]
  u16* sS = (u16*)(smem + 16384);     // [64][128] (p-major)
  float* sAc = (float*)(smem + 32768);
  int bc = blockIdx.x;
  int tid = threadIdx.x, lane = tid & 31, w = tid >> 5;
  size_t rowbase = (size_t)bc * CHUNK;
  for (int i = tid; i < CHUNK * D_STATE; i += 256) sC[i] = Cm[rowbase * D_STATE + i];
  __syncthreads();
  for (int h = 0; h < NHEADS; ++h) {
    if (tid < 64) sAc[tid] = Ac[((size_t)bc * NHEADS + h) * CHUNK + tid];
    for (int i = tid; i < HEADDIM * D_STATE; i += 256)
      sS[i] = Sprev[((size_t)bc * NHEADS + h) * (HEADDIM * D_STATE) + i];
    __syncthreads();
#pragma unroll
    for (int s2 = 0; s2 < 2; ++s2) {
      int st = w + s2 * 8, mt = st & 3, nt = st >> 2;
      v8f acc = {};
      for (int k0 = 0; k0 < D_STATE; k0 += 32) {
        v16bf af = frag_a_gen(sC + mt * 16 * D_STATE + k0, D_STATE, 1);
        v16bf bf = frag_b_gen(sS + (nt * 16) * D_STATE + k0, 1, D_STATE);
        acc = wmma_bf16(af, bf, acc);
      }
      int n = nt * 16 + (lane & 15), mh = mt * 16 + (lane >> 4) * 8;
#pragma unroll
      for (int r = 0; r < 8; ++r) {
        int l = mh + r;
        size_t yi = ((rowbase + l) * NHEADS + h) * HEADDIM + n;
        Y[yi] += expf(sAc[l]) * acc[r];
      }
    }
    __syncthreads();
  }
}

__global__ __launch_bounds__(256) void k_gate_norm(const float* __restrict__ Y,
                                                   const float* __restrict__ zx,
                                                   const float* __restrict__ wn,
                                                   u16* __restrict__ Yg) {
  int row = blockIdx.x, tid = threadIdx.x;
  __shared__ float red[256];
  float v[8]; float ss = 0.f;
#pragma unroll
  for (int i = 0; i < 8; ++i) {
    int d = tid + i * 256;
    float y = Y[(size_t)row * D_INNER + d];
    float z = zx[(size_t)row * D_IN_PROJ + d];
    float t = y * (z / (1.f + expf(-z)));
    v[i] = t; ss += t * t;
  }
  red[tid] = ss; __syncthreads();
  for (int s = 128; s > 0; s >>= 1) {
    if (tid < s) red[tid] += red[tid + s];
    __syncthreads();
  }
  float sc = rsqrtf(red[0] / (float)D_INNER + EPSF);
#pragma unroll
  for (int i = 0; i < 8; ++i) {
    int d = tid + i * 256;
    Yg[(size_t)row * D_INNER + d] = f2bf(v[i] * sc * wn[d]);
  }
}

extern "C" void kernel_launch(void* const* d_in, const int* in_sizes, int n_in,
                              void* d_out, int out_size, void* d_ws, size_t ws_size,
                              hipStream_t stream) {
  const float* x        = (const float*)d_in[0];
  const float* norm_w   = (const float*)d_in[1];
  const float* Wi       = (const float*)d_in[2];
  const float* conv_w   = (const float*)d_in[3];
  const float* conv_b   = (const float*)d_in[4];
  const float* dt_bias  = (const float*)d_in[5];
  const float* A_log    = (const float*)d_in[6];
  const float* Dp       = (const float*)d_in[7];
  const float* norm_in  = (const float*)d_in[8];
  const float* Wo       = (const float*)d_in[9];

  char* ws = (char*)d_ws;
  u16*   xn  = (u16*)(ws + OFF_XN);
  u16*   wib = (u16*)(ws + OFF_WI);
  u16*   wob = (u16*)(ws + OFF_WO);
  float* zx  = (float*)(ws + OFF_ZX);
  float* dtb = (float*)(ws + OFF_DT);
  float* Acb = (float*)(ws + OFF_AC);
  float* Alb = (float*)(ws + OFF_AL);
  u16*   xs  = (u16*)(ws + OFF_XS);
  u16*   xd  = (u16*)(ws + OFF_XD);
  u16*   Bmb = (u16*)(ws + OFF_BM);
  u16*   Cmb = (u16*)(ws + OFF_CM);
  u16*   Sl  = (u16*)(ws + OFF_SL);
  u16*   Sp  = (u16*)(ws + OFF_SP);
  float* Yb  = (float*)(ws + OFF_Y);
  u16*   Yg  = (u16*)(ws + OFF_YG);

  k_rmsnorm_cast<<<ROWS, 256, 0, stream>>>(x, norm_w, xn);
  k_cast<<<(D_IN_PROJ * D_MODEL + 255) / 256, 256, 0, stream>>>(Wi, wib, D_IN_PROJ * D_MODEL);
  k_cast<<<(D_MODEL * D_INNER + 255) / 256, 256, 0, stream>>>(Wo, wob, D_MODEL * D_INNER);

  // 256x32 block tile, 8 waves on M, 2x2 WMMA register tile per wave
  k_gemm<<<dim3(D_IN_PROJ / 32, ROWS / 256), 256, 0, stream>>>(
      xn, wib, zx, nullptr, ROWS, D_IN_PROJ, D_MODEL);

  k_dt<<<(ROWS * NHEADS + 255) / 256, 256, 0, stream>>>(zx, dt_bias, dtb);
  k_cumsum<<<BATCH * NCHUNK * NHEADS, 64, 0, stream>>>(dtb, A_log, Acb, Alb);
  k_conv<<<ROWS, 256, 0, stream>>>(zx, conv_w, conv_b, dtb, xs, xd, Bmb, Cmb);

  k_ssd_intra<<<BATCH * NCHUNK, 256, 0, stream>>>(Bmb, Cmb, xd, xs, Acb, Alb, Dp, Yb, Sl);
  k_ssd_scan<<<BATCH * NHEADS, 256, 0, stream>>>(Sl, Alb, Sp);
  k_ssd_inter<<<BATCH * NCHUNK, 256, 0, stream>>>(Cmb, Sp, Acb, Yb);

  k_gate_norm<<<ROWS, 256, 0, stream>>>(Yb, zx, norm_in, Yg);
  k_gemm<<<dim3(D_MODEL / 32, ROWS / 256), 256, 0, stream>>>(
      Yg, wob, (float*)d_out, x, ROWS, D_MODEL, D_INNER);
}
